// TransformerBlock_8727373545564
// MI455X (gfx1250) — matrix operand order repaired and checked by reading, not compiled, not run on hardware
//
#include <hip/hip_runtime.h>
#include <hip/hip_bf16.h>

// ---------------------------------------------------------------------------
// CDNA5 (gfx1250, wave32) transformer block:
//   bf16 WMMA GEMMs (128x128 block tile, LDS double-buffered via
//   GLOBAL_LOAD_ASYNC_TO_LDS_B128 / ASYNCcnt) + causal flash attention.
// ---------------------------------------------------------------------------

typedef __attribute__((ext_vector_type(8)))  __bf16 v8bf;
typedef __attribute__((ext_vector_type(16))) __bf16 v16bf;
typedef __attribute__((ext_vector_type(8)))  float  v8f;

#define D_MODEL 1024
#define SEQ     2048
#define BATCHN  2
#define HEADS   16
#define DKH     64
#define D_FF    4096
#define MROWS   (BATCHN * SEQ)   // 4096

static __device__ __forceinline__ __bf16 to_bf16(float f) {
    unsigned u = __builtin_bit_cast(unsigned, f);
    u += 0x7fffu + ((u >> 16) & 1u);              // round-to-nearest-even
    unsigned short h = (unsigned short)(u >> 16);
    return __builtin_bit_cast(__bf16, h);
}

static __device__ __forceinline__ v16bf cat16(v8bf lo, v8bf hi) {
    return __builtin_shufflevector(lo, hi, 0,1,2,3,4,5,6,7,8,9,10,11,12,13,14,15);
}

static __device__ __forceinline__ v8f wmma_bf16(v16bf a, v16bf b, v8f c) {
    // (neg_a, A, neg_b, B, c_mod, C, reuse_a, reuse_b)
    return __builtin_amdgcn_wmma_f32_16x16x32_bf16(false, a, false, b, (short)0, c,
                                                   false, false);
}

static __device__ __forceinline__ v8f vzero8() {
    v8f z;
#pragma unroll
    for (int e = 0; e < 8; ++e) z[e] = 0.0f;
    return z;
}

// Async DMA: global memory -> LDS, 16 bytes per lane, tracked by ASYNCcnt.
static __device__ __forceinline__ void async_ld_b128(void* lds, const void* g) {
    unsigned l = (unsigned)(unsigned long long)lds;       // LDS byte address
    unsigned long long ga = (unsigned long long)g;        // 64-bit global addr
    asm volatile("global_load_async_to_lds_b128 %0, %1, off"
                 :: "v"(l), "v"(ga) : "memory");
}
static __device__ __forceinline__ void wait_async0() {
    asm volatile("s_wait_asynccnt 0" ::: "memory");
}

// ---------------------------------------------------------------------------
// f32 -> bf16 conversion (weights)
// ---------------------------------------------------------------------------
__global__ __launch_bounds__(256) void cvt_bf16_kernel(const float* __restrict__ in,
                                                       __bf16* __restrict__ out, int n) {
    int i = blockIdx.x * 256 + threadIdx.x;
    if (i < n) out[i] = to_bf16(in[i]);
}

// ---------------------------------------------------------------------------
// LayerNorm over D_MODEL, f32 in -> bf16 out.  One block per row.
// ---------------------------------------------------------------------------
__global__ __launch_bounds__(256) void ln_kernel(const float* __restrict__ x,
                                                 const float* __restrict__ w,
                                                 const float* __restrict__ b,
                                                 __bf16* __restrict__ out) {
    int row  = blockIdx.x;
    int lane = threadIdx.x & 31;
    int wid  = threadIdx.x >> 5;
    const float* xr = x + (size_t)row * D_MODEL;

    float vals[4];
    float s = 0.0f, ss = 0.0f;
#pragma unroll
    for (int e = 0; e < 4; ++e) {
        float v = xr[threadIdx.x + e * 256];
        vals[e] = v;
        s += v;
        ss += v * v;
    }
#pragma unroll
    for (int m = 16; m >= 1; m >>= 1) {
        s  += __shfl_xor(s,  m, 32);
        ss += __shfl_xor(ss, m, 32);
    }
    __shared__ float rs[8], rss[8];
    if (lane == 0) { rs[wid] = s; rss[wid] = ss; }
    __syncthreads();
    float ts = 0.0f, tss = 0.0f;
#pragma unroll
    for (int k = 0; k < 8; ++k) { ts += rs[k]; tss += rss[k]; }
    float mean = ts * (1.0f / D_MODEL);
    float var  = tss * (1.0f / D_MODEL) - mean * mean;
    float rstd = rsqrtf(var + 1e-5f);
#pragma unroll
    for (int e = 0; e < 4; ++e) {
        int c = threadIdx.x + e * 256;
        out[(size_t)row * D_MODEL + c] = to_bf16((vals[e] - mean) * rstd * w[c] + b[c]);
    }
}

// ---------------------------------------------------------------------------
// WMMA GEMM: C[M,N] = A[M,K](bf16) @ W[N,K](bf16)^T
// Block = 256 threads (8 wave32s), block tile 128(M) x 128(N), K slabs of 64,
// LDS double-buffered: slab k+1 staged by async DMA while slab k is computed.
// Wave (wm = wid&3, wn = wid>>2) owns 32(M) x 64(N) = 8 accumulators,
// 16 WMMAs per K slab.
// Epilogue modes:
//   0: f32 row-major           2: f32 = resid + C (row-major)
//   3: relu -> bf16 row-major  4: bf16 scatter -> [B,H,S,dk]
//   5: bf16 scatter -> [B,H,dk,S]   (transposed V for the P@V WMMA)
// ---------------------------------------------------------------------------
__global__ __launch_bounds__(256) void gemm_bf16_kernel(const __bf16* __restrict__ A,
                                                        const __bf16* __restrict__ W,
                                                        int Mdim, int Ndim, int Kdim,
                                                        int mode,
                                                        const float* __restrict__ resid,
                                                        float* __restrict__ outF,
                                                        __bf16* __restrict__ outB) {
    __shared__ __align__(16) __bf16 As[2][128][72];   // 128 x 64 (+pad), x2 buffers
    __shared__ __align__(16) __bf16 Ws[2][128][72];

    int lane = threadIdx.x & 31;
    int wid  = threadIdx.x >> 5;
    int wm = wid & 3, wn = wid >> 2;
    int hh = lane >> 4, ln = lane & 15;
    int bm = blockIdx.y * 128;
    int bn = blockIdx.x * 128;

    v8f acc[2][4];
#pragma unroll
    for (int u = 0; u < 2; ++u)
#pragma unroll
        for (int j = 0; j < 4; ++j) acc[u][j] = vzero8();

    // stage one 128x64 A slab + 128x64 W slab into buffer `buf` (async DMA)
    auto stage = [&](int buf, int k0) {
#pragma unroll
        for (int t = 0; t < 4; ++t) {
            int task = threadIdx.x + t * 256;     // 1024 b128 segments each
            int r = task >> 3, sg = task & 7;
            async_ld_b128(&As[buf][r][sg * 8],
                          A + (size_t)(bm + r) * Kdim + k0 + sg * 8);
            async_ld_b128(&Ws[buf][r][sg * 8],
                          W + (size_t)(bn + r) * Kdim + k0 + sg * 8);
        }
    };

    int nk = Kdim >> 6;
    stage(0, 0);
    wait_async0();
    __syncthreads();

    int cur = 0;
    for (int it = 0; it < nk; ++it) {
        if (it + 1 < nk) stage(cur ^ 1, (it + 1) << 6);   // DMA next slab

        // compute current slab: 2 K-chunks of 32
#pragma unroll
        for (int c = 0; c < 2; ++c) {
            v16bf af[2];
#pragma unroll
            for (int u = 0; u < 2; ++u) {
                const __bf16* ar = &As[cur][wm * 32 + u * 16 + ln][c * 32 + hh * 8];
                af[u] = cat16(*(const v8bf*)ar, *(const v8bf*)(ar + 16));
            }
#pragma unroll
            for (int j = 0; j < 4; ++j) {
                const __bf16* wr = &Ws[cur][wn * 64 + j * 16 + ln][c * 32 + hh * 8];
                v16bf bf = cat16(*(const v8bf*)wr, *(const v8bf*)(wr + 16));
#pragma unroll
                for (int u = 0; u < 2; ++u)
                    acc[u][j] = wmma_bf16(af[u], bf, acc[u][j]);
            }
        }

        wait_async0();        // next slab fully resident in LDS
        __syncthreads();      // all waves done reading `cur` before overwrite
        cur ^= 1;
    }

    // epilogue: D layout = VGPR i, lanes 0-15 -> M=i, lanes 16-31 -> M=i+8
#pragma unroll
    for (int u = 0; u < 2; ++u) {
#pragma unroll
        for (int j = 0; j < 4; ++j) {
#pragma unroll
            for (int i = 0; i < 8; ++i) {
                int m = bm + wm * 32 + u * 16 + i + 8 * hh;
                int n = bn + wn * 64 + j * 16 + ln;
                float v = acc[u][j][i];
                if (mode == 2) {
                    size_t idx = (size_t)m * Ndim + n;
                    outF[idx] = resid[idx] + v;
                } else if (mode == 3) {
                    outB[(size_t)m * Ndim + n] = to_bf16(fmaxf(v, 0.0f));
                } else if (mode == 4) {          // [B*S, H*dk] -> [B,H,S,dk]
                    int b = m >> 11, s = m & 2047;
                    int h = n >> 6,  d = n & 63;
                    outB[(((size_t)(b * HEADS + h)) * SEQ + s) * DKH + d] = to_bf16(v);
                } else if (mode == 5) {          // [B*S, H*dk] -> [B,H,dk,S]
                    int b = m >> 11, s = m & 2047;
                    int h = n >> 6,  d = n & 63;
                    outB[(((size_t)(b * HEADS + h)) * DKH + d) * SEQ + s] = to_bf16(v);
                } else {
                    outF[(size_t)m * Ndim + n] = v;
                }
            }
        }
    }
}

// ---------------------------------------------------------------------------
// Causal flash attention, bf16 WMMA, dk = 64, scale = 1/8.
// Grid: (SEQ/128, B*H).  Block = 256 threads; each wave owns 16 query rows.
// Per 64-key block: K/V tiles async-DMAed to LDS, 8 WMMAs for Q@K^T, online
// softmax, P relaid through per-wave LDS into A-layout, 8 WMMAs for P@V
// (V pre-transposed [dk,S]).
// ---------------------------------------------------------------------------
__global__ __launch_bounds__(256) void attn_kernel(const __bf16* __restrict__ Q,
                                                   const __bf16* __restrict__ K,
                                                   const __bf16* __restrict__ Vt,
                                                   __bf16* __restrict__ O) {
    __shared__ __align__(16) __bf16 Ks[64][72];
    __shared__ __align__(16) __bf16 Vs[64][72];
    __shared__ __align__(16) __bf16 Ps[8][16][72];

    int lane = threadIdx.x & 31;
    int wid  = threadIdx.x >> 5;
    int hh = lane >> 4, ln = lane & 15;
    int bh   = blockIdx.y;
    int qblk = blockIdx.x;
    int q0   = qblk * 128 + wid * 16;

    const __bf16* Qb = Q  + (size_t)bh * SEQ * DKH;
    const __bf16* Kb = K  + (size_t)bh * SEQ * DKH;
    const __bf16* Vb = Vt + (size_t)bh * DKH * SEQ;

    // Q fragments for the wave's 16 rows (K-dim 64 -> 2 chunks of 32)
    v16bf qa[2];
    const __bf16* qrow = Qb + (size_t)(q0 + ln) * DKH;
#pragma unroll
    for (int c = 0; c < 2; ++c) {
        const v8bf* p0 = (const v8bf*)(qrow + c * 32 + hh * 8);
        const v8bf* p1 = (const v8bf*)(qrow + c * 32 + 16 + hh * 8);
        qa[c] = cat16(*p0, *p1);
    }

    v8f o[4];
#pragma unroll
    for (int t = 0; t < 4; ++t) o[t] = vzero8();
    float mrow[8], lrow[8];
#pragma unroll
    for (int i = 0; i < 8; ++i) { mrow[i] = -1.0e30f; lrow[i] = 0.0f; }

    int sr = threadIdx.x >> 2;        // staging row 0..63
    int sg = threadIdx.x & 3;         // 16-half segment 0..3
    int nkb = qblk * 2 + 2;           // causal: key blocks covering k <= qmax

    for (int kb = 0; kb < nkb; ++kb) {
        int k0 = kb * 64;
        __syncthreads();
        // async stage K tile [64 keys x 64 d] and V^T tile [64 d x 64 keys]
        async_ld_b128(&Ks[sr][sg * 16],     Kb + (size_t)(k0 + sr) * DKH + sg * 16);
        async_ld_b128(&Ks[sr][sg * 16 + 8], Kb + (size_t)(k0 + sr) * DKH + sg * 16 + 8);
        async_ld_b128(&Vs[sr][sg * 16],     Vb + (size_t)sr * SEQ + k0 + sg * 16);
        async_ld_b128(&Vs[sr][sg * 16 + 8], Vb + (size_t)sr * SEQ + k0 + sg * 16 + 8);
        wait_async0();
        __syncthreads();

        // scores S = Q @ K^T : 4 key subtiles x 2 K-chunks
        v8f sacc[4];
#pragma unroll
        for (int j = 0; j < 4; ++j) sacc[j] = vzero8();
#pragma unroll
        for (int j = 0; j < 4; ++j) {
#pragma unroll
            for (int c = 0; c < 2; ++c) {
                const __bf16* kr = &Ks[j * 16 + ln][c * 32 + hh * 8];
                v16bf bb = cat16(*(const v8bf*)kr, *(const v8bf*)(kr + 16));
                sacc[j] = wmma_bf16(qa[c], bb, sacc[j]);
            }
        }

        // online softmax, row = (i + 8*hh) of this wave
#pragma unroll
        for (int i = 0; i < 8; ++i) {
            int q = q0 + i + 8 * hh;
            float sv[4];
            float rm = -1.0e30f;
#pragma unroll
            for (int j = 0; j < 4; ++j) {
                int kcol = k0 + j * 16 + ln;
                float v = (float)sacc[j][i] * 0.125f;      // 1/sqrt(64)
                v = (kcol <= q) ? v : -1.0e30f;            // causal mask
                sv[j] = v;
                rm = fmaxf(rm, v);
            }
#pragma unroll
            for (int m = 8; m >= 1; m >>= 1) rm = fmaxf(rm, __shfl_xor(rm, m, 32));
            float mnew = fmaxf(mrow[i], rm);
            float corr = __expf(mrow[i] - mnew);
            float rs2 = 0.0f;
#pragma unroll
            for (int j = 0; j < 4; ++j) {
                float p = __expf(sv[j] - mnew);
                rs2 += p;
                Ps[wid][i + 8 * hh][j * 16 + ln] = to_bf16(p);
            }
#pragma unroll
            for (int m = 8; m >= 1; m >>= 1) rs2 += __shfl_xor(rs2, m, 32);
            lrow[i] = lrow[i] * corr + rs2;
            mrow[i] = mnew;
#pragma unroll
            for (int t = 0; t < 4; ++t) o[t][i] *= corr;
        }

        // O += P @ V   (A = P from per-wave LDS, B = V^T tile)
#pragma unroll
        for (int c = 0; c < 2; ++c) {
            const __bf16* pr = &Ps[wid][ln][c * 32 + hh * 8];
            v16bf pa = cat16(*(const v8bf*)pr, *(const v8bf*)(pr + 16));
#pragma unroll
            for (int t = 0; t < 4; ++t) {
                const __bf16* vr = &Vs[t * 16 + ln][c * 32 + hh * 8];
                v16bf vb = cat16(*(const v8bf*)vr, *(const v8bf*)(vr + 16));
                o[t] = wmma_bf16(pa, vb, o[t]);
            }
        }
    }

    // write O back to row-major [B*S, D_MODEL] bf16 (head h columns)
    int b = bh >> 4, h = bh & 15;
#pragma unroll
    for (int i = 0; i < 8; ++i) {
        int q = q0 + i + 8 * hh;
        float inv = 1.0f / lrow[i];
#pragma unroll
        for (int t = 0; t < 4; ++t) {
            O[((size_t)(b * SEQ + q)) * D_MODEL + h * DKH + t * 16 + ln] =
                to_bf16(o[t][i] * inv);
        }
    }
}

// ---------------------------------------------------------------------------
// host-side orchestration
// ---------------------------------------------------------------------------
extern "C" void kernel_launch(void* const* d_in, const int* in_sizes, int n_in,
                              void* d_out, int out_size, void* d_ws, size_t ws_size,
                              hipStream_t stream) {
    (void)in_sizes; (void)n_in; (void)out_size; (void)ws_size;

    const float* x     = (const float*)d_in[0];
    const float* W_q   = (const float*)d_in[1];
    const float* W_k   = (const float*)d_in[2];
    const float* W_v   = (const float*)d_in[3];
    const float* W_o   = (const float*)d_in[4];
    const float* W1    = (const float*)d_in[5];
    const float* W2    = (const float*)d_in[6];
    const float* ln1_w = (const float*)d_in[7];
    const float* ln1_b = (const float*)d_in[8];
    const float* ln2_w = (const float*)d_in[9];
    const float* ln2_b = (const float*)d_in[10];
    float* out = (float*)d_out;

    // workspace carve (256B aligned)
    char* p = (char*)d_ws;
    auto carve = [&](size_t bytes) -> void* {
        void* r = (void*)p;
        p += (bytes + 255) & ~(size_t)255;
        return r;
    };
    const size_t DD = (size_t)D_MODEL * D_MODEL;
    const size_t FD = (size_t)D_FF * D_MODEL;
    __bf16* wq   = (__bf16*)carve(DD * 2);
    __bf16* wk   = (__bf16*)carve(DD * 2);
    __bf16* wv   = (__bf16*)carve(DD * 2);
    __bf16* wo   = (__bf16*)carve(DD * 2);
    __bf16* w1   = (__bf16*)carve(FD * 2);
    __bf16* w2   = (__bf16*)carve(FD * 2);
    __bf16* ln1o = (__bf16*)carve((size_t)MROWS * D_MODEL * 2);
    __bf16* Qb   = (__bf16*)carve((size_t)BATCHN * HEADS * SEQ * DKH * 2);
    __bf16* Kb   = (__bf16*)carve((size_t)BATCHN * HEADS * SEQ * DKH * 2);
    __bf16* Vtb  = (__bf16*)carve((size_t)BATCHN * HEADS * DKH * SEQ * 2);
    __bf16* attO = (__bf16*)carve((size_t)MROWS * D_MODEL * 2);
    float*  x1   = (float*)carve((size_t)MROWS * D_MODEL * 4);
    __bf16* ln2o = (__bf16*)carve((size_t)MROWS * D_MODEL * 2);
    __bf16* hb   = (__bf16*)carve((size_t)MROWS * D_FF * 2);

    // 1) weights -> bf16
    cvt_bf16_kernel<<<(int)(DD / 256), 256, 0, stream>>>(W_q, wq, (int)DD);
    cvt_bf16_kernel<<<(int)(DD / 256), 256, 0, stream>>>(W_k, wk, (int)DD);
    cvt_bf16_kernel<<<(int)(DD / 256), 256, 0, stream>>>(W_v, wv, (int)DD);
    cvt_bf16_kernel<<<(int)(DD / 256), 256, 0, stream>>>(W_o, wo, (int)DD);
    cvt_bf16_kernel<<<(int)(FD / 256), 256, 0, stream>>>(W1, w1, (int)FD);
    cvt_bf16_kernel<<<(int)(FD / 256), 256, 0, stream>>>(W2, w2, (int)FD);

    // 2) LN1
    ln_kernel<<<MROWS, 256, 0, stream>>>(x, ln1_w, ln1_b, ln1o);

    // 3) Q/K/V projections (mode 4: [B,H,S,dk] scatter; mode 5: [B,H,dk,S])
    dim3 gp(D_MODEL / 128, MROWS / 128);
    gemm_bf16_kernel<<<gp, 256, 0, stream>>>(ln1o, wq, MROWS, D_MODEL, D_MODEL, 4,
                                             nullptr, nullptr, Qb);
    gemm_bf16_kernel<<<gp, 256, 0, stream>>>(ln1o, wk, MROWS, D_MODEL, D_MODEL, 4,
                                             nullptr, nullptr, Kb);
    gemm_bf16_kernel<<<gp, 256, 0, stream>>>(ln1o, wv, MROWS, D_MODEL, D_MODEL, 5,
                                             nullptr, nullptr, Vtb);

    // 4) causal flash attention
    dim3 ga(SEQ / 128, BATCHN * HEADS);
    attn_kernel<<<ga, 256, 0, stream>>>(Qb, Kb, Vtb, attO);

    // 5) output projection + residual -> x1 (f32)
    gemm_bf16_kernel<<<gp, 256, 0, stream>>>(attO, wo, MROWS, D_MODEL, D_MODEL, 2,
                                             x, x1, nullptr);

    // 6) LN2
    ln_kernel<<<MROWS, 256, 0, stream>>>(x1, ln2_w, ln2_b, ln2o);

    // 7) FFN up + ReLU -> bf16
    dim3 g1(D_FF / 128, MROWS / 128);
    gemm_bf16_kernel<<<g1, 256, 0, stream>>>(ln2o, w1, MROWS, D_FF, D_MODEL, 3,
                                             nullptr, nullptr, hb);

    // 8) FFN down + residual -> d_out (f32)
    gemm_bf16_kernel<<<gp, 256, 0, stream>>>(hb, w2, MROWS, D_MODEL, D_FF, 2,
                                             x1, out, nullptr);
}
